// Model_29566554865790
// MI455X (gfx1250) — compile-verified, matrix operand order checked by reading, not run on hardware
//
#include <hip/hip_runtime.h>
#include <hip/hip_bf16.h>

// ---------------------------------------------------------------------------
// Autoformer AutoCorrelation for MI455X (gfx1250, wave32, WMMA)
// B=8 L=2048 D=1024 H=16 E=64 TOPK=7
// Pipeline:
//   1) cvt x -> bf16; transpose-convert Wq/Wk/Wv/Wo -> [N,K] bf16
//   2) WMMA bf16 GEMMs: Q,K -> [B,D,L] bf16 (channel-major for correlation),
//      V -> [B*L,D] bf16 row-major (for coalesced delay-gather)
//   3) per-channel circular correlation out of LDS (fp32), 69 GFLOP
//   4) mean over channels, per-batch top-7 + softmax
//   5) delay-aggregate V -> agg bf16
//   6) WMMA GEMM: out = agg @ Wo + bo  (fp32 to d_out)
//   7) tiled transpose: attn = corr^T   (second half of d_out)
// ---------------------------------------------------------------------------

#define Bb    8
#define Ls    2048
#define Dm    1024
#define Hh    16
#define Ee    64
#define TOPK  7
#define Ms    (Bb * Ls)          // 16384 rows
#define Kd    1024               // GEMM K

#define BM    128
#define BN    128
#define BK    32

typedef __attribute__((ext_vector_type(16))) __bf16 v16bf;
typedef __attribute__((ext_vector_type(8)))  float  v8f;
typedef __attribute__((ext_vector_type(4)))  unsigned int u32x4;
typedef __attribute__((ext_vector_type(8)))  int    i32x8;
typedef __attribute__((ext_vector_type(4)))  int    i32x4;

#if defined(__gfx1250__) && __has_builtin(__builtin_amdgcn_tensor_load_to_lds) && __has_builtin(__builtin_amdgcn_s_wait_tensorcnt)
#define USE_TDM 1
#else
#define USE_TDM 0
#endif

// ---------------- helpers ----------------
__device__ __forceinline__ unsigned short f2bf(float f) {
  unsigned u = __float_as_uint(f);
  u += 0x7FFFu + ((u >> 16) & 1u);          // round-to-nearest-even
  return (unsigned short)(u >> 16);
}
__device__ __forceinline__ float bf2f(unsigned short h) {
  return __uint_as_float(((unsigned)h) << 16);
}

#if USE_TDM
// Build a 2D-tile Tensor DMA Descriptor (D#) and issue TENSOR_LOAD_TO_LDS.
// bf16 elements (data_size code 1). All operands are wave-uniform (SGPR).
// This toolchain exposes the 6-arg builtin:
//   (u32x4 g0, i32x8 g1, i32x4 g2, i32x4 g3, i32x8 extra, i32 cpol)
__device__ __forceinline__ void tdm_load_2d(unsigned lds_off, const void* gaddr,
                                            unsigned td0, unsigned td1,
                                            unsigned tile0, unsigned tile1,
                                            unsigned long long stride0) {
  unsigned long long ga = (unsigned long long)gaddr;
  u32x4 g0;
  g0.x = 1u;                                              // count=1, user mode
  g0.y = lds_off;                                         // bits 63:32 lds_addr
  g0.z = (unsigned)(ga & 0xFFFFFFFFu);                    // global_addr lo
  g0.w = (unsigned)((ga >> 32) & 0x01FFFFFFu) | (2u << 30); // addr hi | type=2
  i32x8 g1;
  g1.s0 = (int)(1u << 16);                                // data_size = 2 bytes
  g1.s1 = (int)((td0 & 0xFFFFu) << 16);                   // tensor_dim0[15:0]
  g1.s2 = (int)(((td0 >> 16) & 0xFFFFu) | ((td1 & 0xFFFFu) << 16));
  g1.s3 = (int)(((td1 >> 16) & 0xFFFFu) | ((tile0 & 0xFFFFu) << 16));
  g1.s4 = (int)(tile1 & 0xFFFFu);                         // tile_dim1, tile_dim2=0
  g1.s5 = (int)(stride0 & 0xFFFFFFFFull);                 // dim0 stride lo
  g1.s6 = (int)((stride0 >> 32) & 0xFFFFull);             // dim0 stride hi
  g1.s7 = 0;
  i32x4 z4 = {0, 0, 0, 0};
  i32x8 z8 = {0, 0, 0, 0, 0, 0, 0, 0};
  __builtin_amdgcn_tensor_load_to_lds(g0, g1, z4, z4, z8, 0);
}
#endif

// ---------------- conversion kernels ----------------
__global__ void cvt_x_kernel(const float* __restrict__ src,
                             unsigned short* __restrict__ dst, int n) {
  int i = blockIdx.x * 256 + threadIdx.x;
  if (i < n) dst[i] = f2bf(src[i]);
}

// W [K,N] f32 -> WT [N,K] bf16 (tiled transpose)
__global__ void cvt_wT_kernel(const float* __restrict__ W,
                              unsigned short* __restrict__ WT) {
  __shared__ float tile[32][33];
  int k0 = blockIdx.x * 32, n0 = blockIdx.y * 32;
  int tx = threadIdx.x, ty = threadIdx.y;
  for (int r = ty; r < 32; r += 8)
    tile[r][tx] = W[(size_t)(k0 + r) * Dm + n0 + tx];
  __syncthreads();
  for (int r = ty; r < 32; r += 8)
    WT[(size_t)(n0 + r) * Kd + k0 + tx] = f2bf(tile[tx][r]);
}

// ---------------- WMMA GEMM: C = A[M,K] * BT[N,K]^T + bias ----------------
// mode 0: f32 row-major [M,N]
// mode 1: bf16 channel-major [B, D, L]  (m = b*L + l, n = channel)
// mode 2: bf16 row-major [M,N]
__global__ __launch_bounds__(256)
void gemm_bf16_kernel(const unsigned short* __restrict__ A,
                      const unsigned short* __restrict__ BT,
                      const float* __restrict__ bias,
                      void* __restrict__ Cout,
                      int M, int N, int K, int mode) {
  __shared__ __align__(32) unsigned short As[BM * BK];   // 8 KB
  __shared__ __align__(32) unsigned short Bs[BN * BK];   // 8 KB

  const int tid  = threadIdx.x;
  const int wave = tid >> 5;
  const int lane = tid & 31;
  const int wm   = wave >> 2;          // 0..1 -> 64-row strip
  const int wn   = wave & 3;           // 0..3 -> 32-col strip
  const int m0   = blockIdx.y * BM;
  const int n0   = blockIdx.x * BN;

  v8f acc[4][2];
#pragma unroll
  for (int i = 0; i < 4; ++i)
#pragma unroll
    for (int j = 0; j < 2; ++j) {
      v8f z = {0.f, 0.f, 0.f, 0.f, 0.f, 0.f, 0.f, 0.f};
      acc[i][j] = z;
    }

#if USE_TDM
  const unsigned ldsA_off = (unsigned)(unsigned long long)(&As[0]);
  const unsigned ldsB_off = (unsigned)(unsigned long long)(&Bs[0]);
#endif

  for (int kt = 0; kt < K; kt += BK) {
    __syncthreads();                       // previous compute done
#if USE_TDM
    if (wave == 0) {
      tdm_load_2d(ldsA_off, A  + (size_t)m0 * K + kt, (unsigned)K, 65536u,
                  BK, BM, (unsigned long long)K);
      tdm_load_2d(ldsB_off, BT + (size_t)n0 * K + kt, (unsigned)K, 65536u,
                  BK, BN, (unsigned long long)K);
    }
    __builtin_amdgcn_s_wait_tensorcnt(0);
#else
    {
      int row  = tid >> 1;                 // 0..127
      int colh = (tid & 1) << 4;           // 0 or 16
      const int4* gA = (const int4*)(A + (size_t)(m0 + row) * K + kt + colh);
      int4 a0 = gA[0], a1 = gA[1];
      *(int4*)&As[(row << 5) + colh]     = a0;
      *(int4*)&As[(row << 5) + colh + 8] = a1;
      const int4* gB = (const int4*)(BT + (size_t)(n0 + row) * K + kt + colh);
      int4 b0 = gB[0], b1 = gB[1];
      *(int4*)&Bs[(row << 5) + colh]     = b0;
      *(int4*)&Bs[(row << 5) + colh + 8] = b1;
    }
#endif
    __syncthreads();

    const int lr = lane & 15;
    const int kh = (lane >> 4) << 4;       // 0 or 16 within BK
    v16bf af[4], bfr[2];
#pragma unroll
    for (int i = 0; i < 4; ++i)
      af[i] = *(const v16bf*)&As[((wm * 64 + i * 16 + lr) << 5) + kh];
#pragma unroll
    for (int j = 0; j < 2; ++j)
      bfr[j] = *(const v16bf*)&Bs[((wn * 32 + j * 16 + lr) << 5) + kh];

#pragma unroll
    for (int i = 0; i < 4; ++i)
#pragma unroll
      for (int j = 0; j < 2; ++j)
        acc[i][j] = __builtin_amdgcn_wmma_f32_16x16x32_bf16(
            false, af[i], false, bfr[j], (short)0, acc[i][j], false, false);
  }

  // store: C/D layout -> lanes 0-15: N=lane, M=row r; lanes 16-31: M=row r+8
#pragma unroll
  for (int i = 0; i < 4; ++i)
#pragma unroll
    for (int j = 0; j < 2; ++j) {
      int gn     = n0 + wn * 32 + j * 16 + (lane & 15);
      int gmBase = m0 + wm * 64 + i * 16 + ((lane >> 4) << 3);
      float bv   = bias[gn];
#pragma unroll
      for (int r = 0; r < 8; ++r) {
        int gm = gmBase + r;
        float val = acc[i][j][r] + bv;
        if (mode == 0) {
          ((float*)Cout)[(size_t)gm * N + gn] = val;
        } else if (mode == 1) {
          int b = gm >> 11, l = gm & 2047;
          ((unsigned short*)Cout)[((((size_t)b << 10) + gn) << 11) + l] = f2bf(val);
        } else {
          ((unsigned short*)Cout)[(size_t)gm * N + gn] = f2bf(val);
        }
      }
    }
}

// ---------------- circular cross-correlation ----------------
// corr[c, tau] = sum_t q[c,t] * k[c, (t+tau) mod L], one block per channel.
__global__ __launch_bounds__(256)
void corr_kernel(const unsigned short* __restrict__ qc,
                 const unsigned short* __restrict__ kc,
                 float* __restrict__ corr) {
  __shared__ float qs[Ls];        // 8 KB
  __shared__ float ks[2 * Ls];    // 16 KB (doubled for circular indexing)
  const int c   = blockIdx.x;     // 0 .. B*D-1
  const int tid = threadIdx.x;
  const size_t base = (size_t)c << 11;

#if defined(__gfx1250__)
  __builtin_prefetch(qc + base + Ls, 0, 0);   // global_prefetch_b8 (next channel)
#endif
  for (int t = tid; t < Ls; t += 256) {
    qs[t] = bf2f(qc[base + t]);
    float kv = bf2f(kc[base + t]);
    ks[t] = kv;
    ks[t + Ls] = kv;
  }
  __syncthreads();

  float acc[8] = {0.f, 0.f, 0.f, 0.f, 0.f, 0.f, 0.f, 0.f};
#pragma unroll 4
  for (int t = 0; t < Ls; ++t) {
    float qv = qs[t];
#pragma unroll
    for (int j = 0; j < 8; ++j) acc[j] += qv * ks[t + tid + 256 * j];
  }
#pragma unroll
  for (int j = 0; j < 8; ++j) corr[base + tid + 256 * j] = acc[j];
}

// ---------------- mean over channels: [B,D,L] -> [B,L] ----------------
__global__ __launch_bounds__(256)
void mean_kernel(const float* __restrict__ corr, float* __restrict__ meanv) {
  const int tl  = threadIdx.x & 63;
  const int grp = threadIdx.x >> 6;            // 4 channel groups
  const int b   = blockIdx.y;
  const int tau = blockIdx.x * 64 + tl;
  const float* base = corr + ((size_t)b << 21) + tau;
  float s = 0.f;
  for (int c = grp * 256; c < (grp + 1) * 256; ++c) s += base[(size_t)c << 11];
  __shared__ float red[256];
  red[threadIdx.x] = s;
  __syncthreads();
  if (grp == 0) {
    float t = red[tl] + red[64 + tl] + red[128 + tl] + red[192 + tl];
    meanv[((size_t)b << 11) + tau] = t * (1.0f / 1024.0f);
  }
}

// ---------------- per-batch top-7 + softmax ----------------
__global__ __launch_bounds__(256)
void topk_kernel(const float* __restrict__ meanv,
                 float* __restrict__ wts, int* __restrict__ dly) {
  const int b = blockIdx.x, tid = threadIdx.x;
  __shared__ float vals[Ls];
  __shared__ float rv[256];
  __shared__ int   ri[256];
  __shared__ float topv[TOPK];
  __shared__ int   topi[TOPK];
  for (int t = tid; t < Ls; t += 256) vals[t] = meanv[((size_t)b << 11) + t];
  __syncthreads();
  for (int it = 0; it < TOPK; ++it) {
    float best = -3.4e38f; int bi = 0;
    for (int t = tid; t < Ls; t += 256)
      if (vals[t] > best) { best = vals[t]; bi = t; }
    rv[tid] = best; ri[tid] = bi;
    __syncthreads();
    for (int s2 = 128; s2 > 0; s2 >>= 1) {
      if (tid < s2 && rv[tid + s2] > rv[tid]) { rv[tid] = rv[tid + s2]; ri[tid] = ri[tid + s2]; }
      __syncthreads();
    }
    if (tid == 0) { topv[it] = rv[0]; topi[it] = ri[0]; vals[ri[0]] = -3.4e38f; }
    __syncthreads();
  }
  if (tid == 0) {
    float mx = topv[0], s = 0.f, e[TOPK];
    for (int i = 0; i < TOPK; ++i) { e[i] = __expf(topv[i] - mx); s += e[i]; }
    float inv = 1.0f / s;
    for (int i = 0; i < TOPK; ++i) { wts[b * TOPK + i] = e[i] * inv; dly[b * TOPK + i] = topi[i]; }
  }
}

// ---------------- delay aggregation: agg[b,l,d] = sum_i w_i * v[b,(l+d_i)%L,d]
__global__ __launch_bounds__(256)
void agg_kernel(const unsigned short* __restrict__ v_rm,
                const float* __restrict__ wts, const int* __restrict__ dly,
                unsigned short* __restrict__ agg) {
  const int gid = blockIdx.x * 256 + threadIdx.x;
  const int d = gid & 1023;
  const int l = (gid >> 10) & 2047;
  const int b = gid >> 21;
  float s = 0.f;
#pragma unroll
  for (int i = 0; i < TOPK; ++i) {
    int ll = (l + dly[b * TOPK + i]) & 2047;
    s += wts[b * TOPK + i] * bf2f(v_rm[((((size_t)b << 11) + ll) << 10) + d]);
  }
  agg[gid] = f2bf(s);
}

// ---------------- attn = corr^T : [B,D,L] f32 -> [B,L,D] f32 ----------------
__global__ void attn_transpose_kernel(const float* __restrict__ corr,
                                      float* __restrict__ attn) {
  __shared__ float tile[32][33];
  const int b = blockIdx.z;
  const int l0 = blockIdx.x * 32, d0 = blockIdx.y * 32;
  const int tx = threadIdx.x, ty = threadIdx.y;
  for (int r = ty; r < 32; r += 8)
    tile[r][tx] = corr[(((size_t)b << 10) + d0 + r) * 2048 + l0 + tx];
  __syncthreads();
  for (int r = ty; r < 32; r += 8)
    attn[(((size_t)b << 11) + l0 + r) * 1024 + d0 + tx] = tile[tx][r];
}

// ---------------------------------------------------------------------------
extern "C" void kernel_launch(void* const* d_in, const int* in_sizes, int n_in,
                              void* d_out, int out_size, void* d_ws, size_t ws_size,
                              hipStream_t stream) {
  const float* x  = (const float*)d_in[0];
  const float* Wq = (const float*)d_in[1];
  const float* bq = (const float*)d_in[2];
  const float* Wk = (const float*)d_in[3];
  const float* bk = (const float*)d_in[4];
  const float* Wv = (const float*)d_in[5];
  const float* bv = (const float*)d_in[6];
  const float* Wo = (const float*)d_in[7];
  const float* bo = (const float*)d_in[8];

  float* out  = (float*)d_out;
  float* attn = (float*)d_out + (size_t)Ms * Dm;   // second tuple element

  char* ws = (char*)d_ws;
  size_t off = 0;
  auto carve = [&](size_t bytes) -> void* {
    void* p = ws + off;
    off += (bytes + 255) & ~(size_t)255;
    return p;
  };

  unsigned short* xb   = (unsigned short*)carve((size_t)Ms * Kd * 2);
  unsigned short* WqT  = (unsigned short*)carve((size_t)Dm * Kd * 2);
  unsigned short* WkT  = (unsigned short*)carve((size_t)Dm * Kd * 2);
  unsigned short* WvT  = (unsigned short*)carve((size_t)Dm * Kd * 2);
  unsigned short* WoT  = (unsigned short*)carve((size_t)Dm * Kd * 2);
  unsigned short* qc   = (unsigned short*)carve((size_t)Bb * Dm * Ls * 2);  // [B,D,L]
  unsigned short* kc   = (unsigned short*)carve((size_t)Bb * Dm * Ls * 2);
  unsigned short* vr   = (unsigned short*)carve((size_t)Ms * Dm * 2);       // [M,D]
  float*          corr = (float*)carve((size_t)Bb * Dm * Ls * 4);           // [B,D,L]
  float*          mv   = (float*)carve((size_t)Bb * Ls * 4);
  float*          wts  = (float*)carve((size_t)Bb * 8 * 4);
  int*            dly  = (int*)carve((size_t)Bb * 8 * 4);
  unsigned short* agg  = (unsigned short*)carve((size_t)Ms * Dm * 2);
  (void)ws_size; (void)in_sizes; (void)n_in; (void)out_size;

  const int nX = Ms * Kd;
  cvt_x_kernel<<<(nX + 255) / 256, 256, 0, stream>>>(x, xb, nX);

  dim3 tgrid(32, 32), tblk(32, 8);
  cvt_wT_kernel<<<tgrid, tblk, 0, stream>>>(Wq, WqT);
  cvt_wT_kernel<<<tgrid, tblk, 0, stream>>>(Wk, WkT);
  cvt_wT_kernel<<<tgrid, tblk, 0, stream>>>(Wv, WvT);
  cvt_wT_kernel<<<tgrid, tblk, 0, stream>>>(Wo, WoT);

  dim3 ggrid(Dm / BN, Ms / BM);   // (8, 128)
  gemm_bf16_kernel<<<ggrid, 256, 0, stream>>>(xb, WqT, bq, qc, Ms, Dm, Kd, 1);
  gemm_bf16_kernel<<<ggrid, 256, 0, stream>>>(xb, WkT, bk, kc, Ms, Dm, Kd, 1);
  gemm_bf16_kernel<<<ggrid, 256, 0, stream>>>(xb, WvT, bv, vr, Ms, Dm, Kd, 2);

  corr_kernel<<<Bb * Dm, 256, 0, stream>>>(qc, kc, corr);

  dim3 mgrid(Ls / 64, Bb);
  mean_kernel<<<mgrid, 256, 0, stream>>>(corr, mv);

  topk_kernel<<<Bb, 256, 0, stream>>>(mv, wts, dly);

  agg_kernel<<<(Ms * Dm) / 256, 256, 0, stream>>>(vr, wts, dly, agg);

  gemm_bf16_kernel<<<ggrid, 256, 0, stream>>>(agg, WoT, bo, out, Ms, Dm, Kd, 0);

  dim3 agrid(Ls / 32, Dm / 32, Bb);
  attn_transpose_kernel<<<agrid, tblk, 0, stream>>>(corr, attn);
}
